// MaskedAutoencoderViT_28192165330968
// MI455X (gfx1250) — matrix operand order repaired and checked by reading, not compiled, not run on hardware
//
#include <hip/hip_runtime.h>
#include <hip/hip_bf16.h>
#include <math.h>

#define D 128
#define H 8
#define G 2048
#define B 4
#define HID 512
#define KEEP 512
#define NENC 513
#define NENC_PAD 528
#define EPSV 1e-5f

typedef __attribute__((ext_vector_type(16))) _Float16 v16h;
typedef __attribute__((ext_vector_type(8)))  _Float16 v8h;
typedef __attribute__((ext_vector_type(8)))  float    v8f;

// ---------------------------------------------------------------------------
// argsort by rank counting: ids_restore[i] = rank(noise[i]); ids_shuffle[rank]=i
// ---------------------------------------------------------------------------
__global__ void rank_kernel(const float* __restrict__ noise,
                            int* __restrict__ ids_shuffle,
                            int* __restrict__ ids_restore,
                            float* __restrict__ mask_out) {
  int idx = blockIdx.x * blockDim.x + threadIdx.x;
  if (idx >= B * G) return;
  int b = idx / G, i = idx % G;
  const float* row = noise + (size_t)b * G;
  float ni = row[i];
  int rank = 0;
  for (int j = 0; j < G; ++j) {
    float nj = row[j];
    rank += (nj < ni) || (nj == ni && j < i);
  }
  ids_restore[idx] = rank;
  ids_shuffle[(size_t)b * G + rank] = i;
  mask_out[idx] = (rank < KEEP) ? 0.0f : 1.0f;
}

// ---------------------------------------------------------------------------
// Weight pack: f32 (K,N) row-major -> f16 WMMA B-fragment order.
// Fragment (nt, kc, lane) holds 16 contiguous f16:  pk[j] = W[kc*32 + 16*half + j][nt*16 + lm]
// One thread per output element (K*N total).
// ---------------------------------------------------------------------------
__global__ void pack_w_kernel(const float* __restrict__ W,
                              _Float16* __restrict__ Wp, int K, int N) {
  int o = blockIdx.x * blockDim.x + threadIdx.x;
  if (o >= K * N) return;
  int j    = o & 15;
  int lane = (o >> 4) & 31;
  int KC   = K >> 5;
  int kc   = (o >> 9) % KC;
  int nt   = (o >> 9) / KC;
  int half = lane >> 4, lm = lane & 15;
  int k = kc * 32 + 16 * half + j;
  int n = nt * 16 + lm;
  Wp[o] = (_Float16)W[(size_t)k * N + n];
}

// ---------------------------------------------------------------------------
// Encoder input build (f32 residual stream X).
// ---------------------------------------------------------------------------
__global__ void enc_build_kernel(const float* __restrict__ x,
                                 const float* __restrict__ pos,
                                 const float* __restrict__ cls,
                                 const float* __restrict__ ew,
                                 const float* __restrict__ eb,
                                 const int* __restrict__ ids_shuffle,
                                 float* __restrict__ X) {
  int t = blockIdx.x, d = threadIdx.x;
  int b = t / NENC_PAD, r = t % NENC_PAD;
  float v;
  if (r == 0) v = cls[d];
  else if (r <= KEEP) {
    int g = ids_shuffle[(size_t)b * G + (r - 1)];
    v = x[(size_t)b * G + g] * ew[d] + eb[d] + pos[(size_t)g * D + d];
  } else v = 0.0f;
  X[(size_t)t * D + d] = v;
}

// ---------------------------------------------------------------------------
// LayerNorm over D=128: one wave32 per token, float4 + shfl reduce, f16 out.
// ---------------------------------------------------------------------------
__global__ void ln_kernel(const float* __restrict__ X,
                          const float* __restrict__ gw,
                          const float* __restrict__ gb,
                          _Float16* __restrict__ Yo, int total) {
  int token = blockIdx.x * (blockDim.x >> 5) + (threadIdx.x >> 5);
  if (token >= total) return;
  int lane = threadIdx.x & 31;
  float4 v = ((const float4*)(X + (size_t)token * D))[lane];
  float s = v.x + v.y + v.z + v.w;
  for (int o = 16; o >= 1; o >>= 1) s += __shfl_xor(s, o, 32);
  float mu = s * (1.0f / D);
  float dx = v.x - mu, dy = v.y - mu, dz = v.z - mu, dw = v.w - mu;
  float vs = dx * dx + dy * dy + dz * dz + dw * dw;
  for (int o = 16; o >= 1; o >>= 1) vs += __shfl_xor(vs, o, 32);
  float inv = rsqrtf(vs * (1.0f / D) + EPSV);
  float4 g4 = ((const float4*)gw)[lane];
  float4 b4 = ((const float4*)gb)[lane];
  v8h* orow = (v8h*)(Yo + (size_t)token * D);
  _Float16* op = Yo + (size_t)token * D + lane * 4;
  op[0] = (_Float16)(dx * inv * g4.x + b4.x);
  op[1] = (_Float16)(dy * inv * g4.y + b4.y);
  op[2] = (_Float16)(dz * inv * g4.z + b4.z);
  op[3] = (_Float16)(dw * inv * g4.w + b4.w);
  (void)orow;
}

// ---------------------------------------------------------------------------
// WMMA GEMM:  C[b] = act(A16[b] @ W + bias) (+resid)
// A16: (B,M,K) f16 row-major.  Wp: packed f16 fragments.
// One wave computes a 16x64 strip: 1 A fragment x 4 B fragments -> 4 WMMAs
// per 32-wide K chunk.  4 waves / block.  M%16==0, N%64==0, K%32==0.
// ---------------------------------------------------------------------------
__global__ void gemm_wmma_kernel(const _Float16* __restrict__ A16,
                                 const _Float16* __restrict__ Wp,
                                 const float* __restrict__ bias,
                                 const float* resid,
                                 float* Cf32, _Float16* C16,
                                 int M, int N, int K, int act) {
  int lane = threadIdx.x & 31;
  int wave = threadIdx.x >> 5;
  int half = lane >> 4, lm = lane & 15;
  int tm = (blockIdx.y * 4 + wave) * 16;
  if (tm >= M) return;
  int ntb = blockIdx.x * 4;            // base N tile (of 4)
  int b = blockIdx.z;
  int KC = K >> 5;
  const _Float16* Arow = A16 + (size_t)b * M * K + (size_t)(tm + lm) * K;
  const size_t ntStride = (size_t)KC * 512;   // halves per N tile of fragments

  v8f acc0 = {}, acc1 = {}, acc2 = {}, acc3 = {};
  for (int kc = 0; kc < KC; ++kc) {
    int k0 = kc * 32;
    // A fragment: 16 contiguous k (k = 16*half + j), matching packed B order
    v8h lo = *(const v8h*)(Arow + k0 + 16 * half);
    v8h hi = *(const v8h*)(Arow + k0 + 16 * half + 8);
    v16h av;
#pragma unroll
    for (int i = 0; i < 8; ++i) { av[i] = lo[i]; av[8 + i] = hi[i]; }
    // B fragments: one 32B contiguous load each (pre-packed)
    const _Float16* base = Wp + ((size_t)(ntb * KC + kc) * 32 + lane) * 16;
    v16h b0 = *(const v16h*)(base);
    v16h b1 = *(const v16h*)(base + ntStride);
    v16h b2 = *(const v16h*)(base + 2 * ntStride);
    v16h b3 = *(const v16h*)(base + 3 * ntStride);
    acc0 = __builtin_amdgcn_wmma_f32_16x16x32_f16(false, av, false, b0, (short)0, acc0, false, false);
    acc1 = __builtin_amdgcn_wmma_f32_16x16x32_f16(false, av, false, b1, (short)0, acc1, false, false);
    acc2 = __builtin_amdgcn_wmma_f32_16x16x32_f16(false, av, false, b2, (short)0, acc2, false, false);
    acc3 = __builtin_amdgcn_wmma_f32_16x16x32_f16(false, av, false, b3, (short)0, acc3, false, false);
  }

  v8f accs[4] = {acc0, acc1, acc2, acc3};
#pragma unroll
  for (int t = 0; t < 4; ++t) {
#pragma unroll
    for (int r = 0; r < 8; ++r) {
      int m = tm + r + 8 * half;
      int n = (ntb + t) * 16 + lm;
      size_t off = (size_t)b * M * N + (size_t)m * N + n;
      float v = accs[t][r];
      if (bias) v += bias[n];
      if (act == 1) v = 0.5f * v * (1.0f + erff(v * 0.70710678118f));
      if (resid) v += resid[off];
      if (Cf32) Cf32[off] = v;
      if (C16)  C16[off]  = (_Float16)v;
    }
  }
}

// ---------------------------------------------------------------------------
// Attention: online softmax, hd=16, scale=1/4; f16 in (QKV), f16 out (Y).
// One thread per (b, h, query).
// ---------------------------------------------------------------------------
__global__ void attn_kernel(const _Float16* __restrict__ QKV,
                            _Float16* __restrict__ Y,
                            int Npad, int Nvalid) {
  int idx = blockIdx.x * blockDim.x + threadIdx.x;
  if (idx >= B * H * Npad) return;
  int q = idx % Npad;
  int h = (idx / Npad) % H;
  int b = idx / (Npad * H);
  _Float16* yo = Y + ((size_t)b * Npad + q) * D + h * 16;
  if (q >= Nvalid) {
#pragma unroll
    for (int d = 0; d < 16; ++d) yo[d] = (_Float16)0.0f;
    return;
  }
  const _Float16* base = QKV + (size_t)b * Npad * (3 * D);
  const _Float16* qp = base + (size_t)q * (3 * D) + h * 16;
  float qv[16];
#pragma unroll
  for (int d = 0; d < 16; ++d) qv[d] = (float)qp[d] * 0.25f;  // hd^-0.5
  float m = -1e30f, s = 0.0f, acc[16];
#pragma unroll
  for (int d = 0; d < 16; ++d) acc[d] = 0.0f;
  for (int kk = 0; kk < Nvalid; ++kk) {
    const _Float16* kp = base + (size_t)kk * (3 * D) + D + h * 16;
    const _Float16* vp = kp + D;
    float dot = 0.0f;
#pragma unroll
    for (int d = 0; d < 16; ++d) dot += qv[d] * (float)kp[d];
    float mn = fmaxf(m, dot);
    float corr = __expf(m - mn);
    float p = __expf(dot - mn);
    s = s * corr + p;
#pragma unroll
    for (int d = 0; d < 16; ++d) acc[d] = acc[d] * corr + p * (float)vp[d];
    m = mn;
  }
  float inv = 1.0f / s;
#pragma unroll
  for (int d = 0; d < 16; ++d) yo[d] = (_Float16)(acc[d] * inv);
}

// ---------------------------------------------------------------------------
// Decoder input: unshuffle kept tokens / mask token + pos embed (f32 stream).
// ---------------------------------------------------------------------------
__global__ void dec_build_kernel(const float* __restrict__ Ydec,  // (B,NENC_PAD,D) f32
                                 const float* __restrict__ pos,
                                 const float* __restrict__ mtok,
                                 const int* __restrict__ ids_restore,
                                 float* __restrict__ X) {
  int t = blockIdx.x, d = threadIdx.x;
  int b = t / G, g = t % G;
  int r = ids_restore[(size_t)b * G + g];
  float v = (r < KEEP) ? Ydec[((size_t)b * NENC_PAD + 1 + r) * D + d] : mtok[d];
  X[(size_t)t * D + d] = v + pos[(size_t)g * D + d];
}

__global__ void latent0_kernel(const _Float16* __restrict__ LN16, float* __restrict__ out) {
  int idx = blockIdx.x * blockDim.x + threadIdx.x;
  if (idx >= B * D) return;
  int b = idx / D, d = idx % D;
  out[idx] = (float)LN16[((size_t)b * NENC_PAD) * D + d];
}

__global__ void pred_kernel(const _Float16* __restrict__ LN16,
                            const float* __restrict__ pw,
                            const float* __restrict__ pb,
                            float* __restrict__ pred) {
  int idx = blockIdx.x * blockDim.x + threadIdx.x;
  if (idx >= B * G) return;
  const _Float16* r = LN16 + (size_t)idx * D;
  float s = 0.0f;
#pragma unroll 4
  for (int d = 0; d < D; ++d) s += (float)r[d] * pw[d];
  pred[idx] = s + pb[0];
}

__global__ void zero2_kernel(float* a) {
  if (threadIdx.x < 2) a[threadIdx.x] = 0.0f;
}

__global__ void loss_reduce_kernel(const float* __restrict__ pred,
                                   const float* __restrict__ x,
                                   const float* __restrict__ mask,
                                   float* acc) {
  int idx = blockIdx.x * blockDim.x + threadIdx.x;
  float se = 0.0f, sm = 0.0f;
  if (idx < B * G) {
    float d = pred[idx] - x[idx];
    se = d * d * mask[idx];
    sm = mask[idx];
  }
  for (int o = 16; o >= 1; o >>= 1) {
    se += __shfl_xor(se, o, 32);
    sm += __shfl_xor(sm, o, 32);
  }
  if ((threadIdx.x & 31) == 0) {
    atomicAdd(acc, se);
    atomicAdd(acc + 1, sm);
  }
}

__global__ void loss_final_kernel(const float* acc, float* out) {
  out[0] = acc[0] / acc[1];
}

// ---------------------------------------------------------------------------
extern "C" void kernel_launch(void* const* d_in, const int* in_sizes, int n_in,
                              void* d_out, int out_size, void* d_ws, size_t ws_size,
                              hipStream_t stream) {
  const float* x          = (const float*)d_in[0];
  const float* noise      = (const float*)d_in[1];
  const float* pos_embed  = (const float*)d_in[2];
  const float* cls_token  = (const float*)d_in[3];
  const float* enc_ew     = (const float*)d_in[4];
  const float* enc_eb     = (const float*)d_in[5];
  const float* norm_w     = (const float*)d_in[6];
  const float* norm_b     = (const float*)d_in[7];
  const float* dec_ew     = (const float*)d_in[8];
  const float* dec_eb     = (const float*)d_in[9];
  const float* mask_token = (const float*)d_in[10];
  const float* dec_nw     = (const float*)d_in[11];
  const float* dec_nb     = (const float*)d_in[12];
  const float* pred_w     = (const float*)d_in[13];
  const float* pred_b     = (const float*)d_in[14];
  const float* enc_p[12];
  const float* dec_p[12];
  for (int i = 0; i < 12; ++i) enc_p[i] = (const float*)d_in[15 + i];
  for (int i = 0; i < 12; ++i) dec_p[i] = (const float*)d_in[27 + i];

  float* out      = (float*)d_out;
  float* o_loss   = out;
  float* o_pred   = out + 1;
  float* o_mask   = out + 1 + B * G;
  float* o_latent = out + 1 + 2 * B * G;

  // ---- workspace carve (all chunks multiple-of-256B, base assumed aligned) --
  char* p = (char*)d_ws;
  auto carve = [&](size_t bytes) { char* r = p; p += (bytes + 255) & ~(size_t)255; return r; };
  const size_t SZ = (size_t)B * G * D;                       // 1,048,576 elems
  float*     X    = (float*)carve(SZ * 4);                   // residual stream (f32)
  float*     YF   = (float*)carve(SZ * 4);                   // f32 temp (dec-embed out)
  _Float16*  LN16 = (_Float16*)carve(SZ * 2);
  _Float16*  QKV16= (_Float16*)carve(SZ * 3 * 2);
  _Float16*  Y16  = (_Float16*)carve(SZ * 2);
  _Float16*  FF16 = (_Float16*)carve((size_t)B * G * HID * 2);
  _Float16*  WPK  = (_Float16*)carve((size_t)(8 * 196608 + 16384) * 2); // packed weights
  float*     acc  = (float*)carve(256);
  int* ids_shuffle = (int*)carve((size_t)B * G * 4);
  int* ids_restore = (int*)carve((size_t)B * G * 4);

  // packed-weight offsets (halves): per layer [qkv | proj | fc1 | fc2]
  const size_t LYR = 196608;
  auto wp_enc = [&](int l, int which) {
    const size_t offs[4] = {0, 49152, 65536, 131072};
    return WPK + (size_t)l * LYR + offs[which];
  };
  auto wp_dec = [&](int l, int which) {
    const size_t offs[4] = {0, 49152, 65536, 131072};
    return WPK + 6 * LYR + (size_t)l * LYR + offs[which];
  };
  _Float16* wp_dembed = WPK + 8 * LYR;

  auto pack = [&](const float* W, _Float16* Wp, int K, int N) {
    int total = K * N;
    hipLaunchKernelGGL(pack_w_kernel, dim3((total + 255) / 256), dim3(256), 0, stream,
                       W, Wp, K, N);
  };
  auto gemm = [&](const _Float16* Ap, const _Float16* Wp, const float* biasp,
                  const float* residp, float* Cf, _Float16* Ch,
                  int M, int N, int K, int act) {
    dim3 grid(N / 64, (M + 63) / 64, B);
    hipLaunchKernelGGL(gemm_wmma_kernel, grid, dim3(128), 0, stream,
                       Ap, Wp, biasp, residp, Cf, Ch, M, N, K, act);
  };
  auto layernorm = [&](const float* Xp, const float* gw, const float* gb,
                       _Float16* Yp, int M) {
    int total = B * M;
    hipLaunchKernelGGL(ln_kernel, dim3((total + 7) / 8), dim3(256), 0, stream,
                       Xp, gw, gb, Yp, total);
  };
  auto block = [&](const float* prm[12], _Float16* wq, _Float16* wpj,
                   _Float16* w1, _Float16* w2, int l, int M, int Nvalid) {
    layernorm(X, prm[0] + l * D, prm[1] + l * D, LN16, M);
    gemm(LN16, wq, prm[3] + l * 3 * D, nullptr, nullptr, QKV16, M, 3 * D, D, 0);
    int atot = B * H * M;
    hipLaunchKernelGGL(attn_kernel, dim3((atot + 255) / 256), dim3(256), 0, stream,
                       QKV16, Y16, M, Nvalid);
    gemm(Y16, wpj, prm[5] + l * D, X, X, nullptr, M, D, D, 0);      // +residual
    layernorm(X, prm[6] + l * D, prm[7] + l * D, LN16, M);
    gemm(LN16, w1, prm[9] + l * HID, nullptr, nullptr, FF16, M, HID, D, 1);  // GELU
    gemm(FF16, w2, prm[11] + l * D, X, X, nullptr, M, D, HID, 0);   // +residual
  };

  // 0. pack all weights into WMMA fragment order (f16)
  for (int l = 0; l < 6; ++l) {
    pack(enc_p[2]  + (size_t)l * D * 3 * D, wp_enc(l, 0), D, 3 * D);
    pack(enc_p[4]  + (size_t)l * D * D,     wp_enc(l, 1), D, D);
    pack(enc_p[8]  + (size_t)l * D * HID,   wp_enc(l, 2), D, HID);
    pack(enc_p[10] + (size_t)l * HID * D,   wp_enc(l, 3), HID, D);
  }
  for (int l = 0; l < 2; ++l) {
    pack(dec_p[2]  + (size_t)l * D * 3 * D, wp_dec(l, 0), D, 3 * D);
    pack(dec_p[4]  + (size_t)l * D * D,     wp_dec(l, 1), D, D);
    pack(dec_p[8]  + (size_t)l * D * HID,   wp_dec(l, 2), D, HID);
    pack(dec_p[10] + (size_t)l * HID * D,   wp_dec(l, 3), HID, D);
  }
  pack(dec_ew, wp_dembed, D, D);

  // 1. masking / shuffle indices + mask output
  hipLaunchKernelGGL(rank_kernel, dim3((B * G + 255) / 256), dim3(256), 0, stream,
                     noise, ids_shuffle, ids_restore, o_mask);

  // 2. encoder input (B, NENC_PAD, D) f32
  hipLaunchKernelGGL(enc_build_kernel, dim3(B * NENC_PAD), dim3(D), 0, stream,
                     x, pos_embed, cls_token, enc_ew, enc_eb, ids_shuffle, X);

  // 3. encoder: 6 blocks
  for (int l = 0; l < 6; ++l)
    block(enc_p, wp_enc(l, 0), wp_enc(l, 1), wp_enc(l, 2), wp_enc(l, 3),
          l, NENC_PAD, NENC);

  // 4. latent = LN(h); export latent[:,0]
  layernorm(X, norm_w, norm_b, LN16, NENC_PAD);
  hipLaunchKernelGGL(latent0_kernel, dim3((B * D + 255) / 256), dim3(256), 0, stream,
                     LN16, o_latent);

  // 5. decoder embed -> YF (f32)
  gemm(LN16, wp_dembed, dec_eb, nullptr, YF, nullptr, NENC_PAD, D, D, 0);

  // 6. decoder input: unshuffle + mask token + pos embed -> X (B,G,D)
  hipLaunchKernelGGL(dec_build_kernel, dim3(B * G), dim3(D), 0, stream,
                     YF, pos_embed, mask_token, ids_restore, X);

  // 7. decoder: 2 blocks (full attention over 2048)
  for (int l = 0; l < 2; ++l)
    block(dec_p, wp_dec(l, 0), wp_dec(l, 1), wp_dec(l, 2), wp_dec(l, 3),
          l, G, G);

  // 8. final LN + prediction head
  layernorm(X, dec_nw, dec_nb, LN16, G);
  hipLaunchKernelGGL(pred_kernel, dim3((B * G + 255) / 256), dim3(256), 0, stream,
                     LN16, pred_w, pred_b, o_pred);

  // 9. masked MSE loss
  hipLaunchKernelGGL(zero2_kernel, dim3(1), dim3(32), 0, stream, acc);
  hipLaunchKernelGGL(loss_reduce_kernel, dim3((B * G + 255) / 256), dim3(256), 0, stream,
                     o_pred, x, o_mask, acc);
  hipLaunchKernelGGL(loss_final_kernel, dim3(1), dim3(1), 0, stream, acc, o_loss);
}
